// LinearTransformerOrgMultiModel_28475633172715
// MI455X (gfx1250) — matrix operand order repaired and checked by reading, not compiled, not run on hardware
//
#include <hip/hip_runtime.h>
#include <hip/hip_bf16.h>
#include <math.h>

// ---------------- model constants ----------------
#define D_MODEL 128
#define D_FF    512
#define NUM_L   2
#define BB      512
#define SS      256
#define MROWS   (SS * BB)        // 131072
#define LN_EPS  1e-5f
#define EPS_F   1e-6f

typedef __attribute__((ext_vector_type(16))) __bf16 bf16x16;
typedef __attribute__((ext_vector_type(8)))  float  floatx8;

struct FragBF { union { bf16x16 v; unsigned int w[8]; }; };

__device__ __forceinline__ unsigned short f2bf(float f) {
    union { float f; unsigned int u; } x; x.f = f;
    unsigned int r = x.u + 0x7FFFu + ((x.u >> 16) & 1u);
    return (unsigned short)(r >> 16);
}
__device__ __forceinline__ float bf2f(unsigned short h) {
    union { unsigned int u; float f; } x; x.u = ((unsigned int)h) << 16; return x.f;
}

// LDS tile row stride (u16 units): 32 data + 8 pad -> row base 80B, 4B-aligned
#define ATS 40

// A-frag 16x32 bf16: lanes 0-15: K0-15; lanes 16-31: K16-31
__device__ __forceinline__ void load_a_frag(FragBF& f, const unsigned short* rowbase, int hi) {
#pragma unroll
    for (int p = 0; p < 8; ++p) f.w[p] = *(const unsigned int*)(rowbase + hi * 16 + 2 * p);
}
// B-frag 32x16 bf16: lane holds fixed column, lanes 0-15: K0-15, lanes 16-31: K16-31
__device__ __forceinline__ void load_b_frag(FragBF& f, const unsigned short* rowbase, int hi) {
#pragma unroll
    for (int p = 0; p < 8; ++p) f.w[p] = *(const unsigned int*)(rowbase + hi * 16 + 2 * p);
}

// =====================================================================
// Generic GEMM:  Y[M,N] = act(A[M,K] @ W[N,K]^T + bias),  bf16 in/out, f32 acc
// grid.x = M/128, grid.y = N/128.  mode: 0=none, 1=relu+1e-6 (Q/K), 2=relu (FF1)
// =====================================================================
__global__ __launch_bounds__(256)
void gemm_bias_act(const unsigned short* __restrict__ A,
                   const unsigned short* __restrict__ W,
                   const float* __restrict__ bias,
                   unsigned short* __restrict__ out,
                   int K, int Nld, int mode) {
    __shared__ unsigned short ldsA[128 * ATS];
    __shared__ unsigned short ldsW[128 * ATS];
    const int m0    = blockIdx.x * 128;
    const int nbase = blockIdx.y * 128;
    const int tid = threadIdx.x, lane = tid & 31, wave = tid >> 5;
    const int r = lane & 15, hi = lane >> 4;

    floatx8 acc[8];
#pragma unroll
    for (int i = 0; i < 8; ++i) {
#pragma unroll
        for (int j = 0; j < 8; ++j) acc[i][j] = 0.0f;
    }

    const int kt_max = K >> 5;
    for (int kt = 0; kt < kt_max; ++kt) {
        const int k0 = kt << 5;
        __syncthreads();
        {   // cooperative staging: 256 thr, each 32B of A and W
            int rowa = tid >> 1, half = tid & 1;
            const unsigned short* ga = A + (size_t)(m0 + rowa) * K + k0 + half * 16;
            uint4 a0 = *(const uint4*)ga, a1 = *(const uint4*)(ga + 8);
            uint4* la = (uint4*)&ldsA[rowa * ATS + half * 16];
            la[0] = a0; la[1] = a1;
            const unsigned short* gw = W + (size_t)(nbase + rowa) * K + k0 + half * 16;
            uint4 w0 = *(const uint4*)gw, w1 = *(const uint4*)(gw + 8);
            uint4* lw = (uint4*)&ldsW[rowa * ATS + half * 16];
            lw[0] = w0; lw[1] = w1;
        }
        __syncthreads();

        FragBF af; load_a_frag(af, &ldsA[(wave * 16 + r) * ATS], hi);
#pragma unroll
        for (int nt = 0; nt < 8; ++nt) {
            FragBF bf; load_b_frag(bf, &ldsW[(nt * 16 + r) * ATS], hi);
            acc[nt] = __builtin_amdgcn_wmma_f32_16x16x32_bf16(
                false, af.v, false, bf.v, (short)0, acc[nt], false, false);
        }
    }
    // epilogue: C/D layout -> VGPR j holds rows j (lanes 0-15) and j+8 (lanes 16-31)
#pragma unroll
    for (int j = 0; j < 8; ++j) {
        int row = m0 + wave * 16 + j + hi * 8;
#pragma unroll
        for (int nt = 0; nt < 8; ++nt) {
            int col = nbase + nt * 16 + r;
            float v = acc[nt][j] + bias[col];
            if (mode == 1)      v = fmaxf(v, 0.0f) + EPS_F;
            else if (mode == 2) v = fmaxf(v, 0.0f);
            out[(size_t)row * Nld + col] = f2bf(v);
        }
    }
}

// =====================================================================
// GEMM (N=128) + optional per-row Z-scale + bias + residual + LayerNorm.
// Updates x in f32 and bf16. W is [N,K]; if per_s, W indexed by s = row/512.
// =====================================================================
__global__ __launch_bounds__(256)
void gemm_ln(const unsigned short* __restrict__ A,
             const unsigned short* __restrict__ Wbase,
             const float* __restrict__ bias,     // may be null
             const float* __restrict__ zscale,   // may be null
             const float* __restrict__ gamma,
             const float* __restrict__ beta,
             float* __restrict__ xf,
             unsigned short* __restrict__ xb,
             int K, int per_s) {
    __shared__ unsigned short ldsA[128 * ATS];
    __shared__ unsigned short ldsW[128 * ATS];
    const int m0 = blockIdx.x * 128;
    const unsigned short* W = per_s ? (Wbase + (size_t)(m0 >> 9) * 128 * K) : Wbase;
    const int tid = threadIdx.x, lane = tid & 31, wave = tid >> 5;
    const int r = lane & 15, hi = lane >> 4;

    floatx8 acc[8];
#pragma unroll
    for (int i = 0; i < 8; ++i) {
#pragma unroll
        for (int j = 0; j < 8; ++j) acc[i][j] = 0.0f;
    }

    const int kt_max = K >> 5;
    for (int kt = 0; kt < kt_max; ++kt) {
        const int k0 = kt << 5;
        __syncthreads();
        {
            int rowa = tid >> 1, half = tid & 1;
            const unsigned short* ga = A + (size_t)(m0 + rowa) * K + k0 + half * 16;
            uint4 a0 = *(const uint4*)ga, a1 = *(const uint4*)(ga + 8);
            uint4* la = (uint4*)&ldsA[rowa * ATS + half * 16];
            la[0] = a0; la[1] = a1;
            const unsigned short* gw = W + (size_t)rowa * K + k0 + half * 16;
            uint4 w0 = *(const uint4*)gw, w1 = *(const uint4*)(gw + 8);
            uint4* lw = (uint4*)&ldsW[rowa * ATS + half * 16];
            lw[0] = w0; lw[1] = w1;
        }
        __syncthreads();

        FragBF af; load_a_frag(af, &ldsA[(wave * 16 + r) * ATS], hi);
#pragma unroll
        for (int nt = 0; nt < 8; ++nt) {
            FragBF bf; load_b_frag(bf, &ldsW[(nt * 16 + r) * ATS], hi);
            acc[nt] = __builtin_amdgcn_wmma_f32_16x16x32_bf16(
                false, af.v, false, bf.v, (short)0, acc[nt], false, false);
        }
    }
    // fused epilogue: y -> t = x + y -> LayerNorm over the 128 cols of each row.
    // A row's 128 values live in 8 frags x 16 lanes of one half-wave, so
    // mean/var reduce = 8 local adds + 4 shfl_xor steps within the 16-lane group.
#pragma unroll
    for (int j = 0; j < 8; ++j) {
        int row = m0 + wave * 16 + j + hi * 8;
        float zs = zscale ? zscale[row] : 1.0f;
        float tv[8];
        float s1 = 0.0f, s2 = 0.0f;
#pragma unroll
        for (int nt = 0; nt < 8; ++nt) {
            int col = nt * 16 + r;
            float v = acc[nt][j] * zs;
            if (bias) v += bias[col];
            float t = xf[(size_t)row * D_MODEL + col] + v;
            tv[nt] = t; s1 += t; s2 += t * t;
        }
#pragma unroll
        for (int off = 1; off < 16; off <<= 1) {
            s1 += __shfl_xor(s1, off, 32);
            s2 += __shfl_xor(s2, off, 32);
        }
        float mu  = s1 * (1.0f / 128.0f);
        float var = s2 * (1.0f / 128.0f) - mu * mu;
        float inv = rsqrtf(var + LN_EPS);
#pragma unroll
        for (int nt = 0; nt < 8; ++nt) {
            int col = nt * 16 + r;
            float o = (tv[nt] - mu) * inv * gamma[col] + beta[col];
            xf[(size_t)row * D_MODEL + col] = o;
            xb[(size_t)row * D_MODEL + col] = f2bf(o);
        }
    }
}

// =====================================================================
// Per-s KV = K_s^T @ V_s  (M=N=128, reduction over batch, K=512).
// K and V staged TRANSPOSED in LDS so both frag types use 32-bit ds loads.
// Result stored transposed [s][e][d] = [N][K] so attn GEMM reuses weight path.
// =====================================================================
__global__ __launch_bounds__(256)
void kv_kernel(const unsigned short* __restrict__ Kb,
               const unsigned short* __restrict__ Vb,
               unsigned short* __restrict__ kvT) {
    __shared__ unsigned short ldsKT[128 * ATS];   // [d][n]
    __shared__ unsigned short ldsVT[128 * ATS];   // [e][n]
    const int s = blockIdx.x;
    const int tid = threadIdx.x, lane = tid & 31, wave = tid >> 5;
    const int r = lane & 15, hi = lane >> 4;

    floatx8 acc[8];
#pragma unroll
    for (int i = 0; i < 8; ++i) {
#pragma unroll
        for (int j = 0; j < 8; ++j) acc[i][j] = 0.0f;
    }

    for (int nc = 0; nc < (BB >> 5); ++nc) {      // 16 chunks of 32 batch rows
        const int n0 = nc << 5;
        __syncthreads();
        {   // transpose-stage: thread -> (n, 16-wide d block)
            int n = tid >> 3;
            int dblk = (tid & 7) * 16;
            const unsigned short* gk = Kb + ((size_t)s * BB + n0 + n) * D_MODEL + dblk;
            const unsigned short* gv = Vb + ((size_t)s * BB + n0 + n) * D_MODEL + dblk;
            union { uint4 q[2]; unsigned short u[16]; } tk, tv;
            tk.q[0] = *(const uint4*)gk; tk.q[1] = *(const uint4*)(gk + 8);
            tv.q[0] = *(const uint4*)gv; tv.q[1] = *(const uint4*)(gv + 8);
#pragma unroll
            for (int i = 0; i < 16; ++i) {
                ldsKT[(dblk + i) * ATS + n] = tk.u[i];
                ldsVT[(dblk + i) * ATS + n] = tv.u[i];
            }
        }
        __syncthreads();

        FragBF af; load_a_frag(af, &ldsKT[(wave * 16 + r) * ATS], hi);   // A[d][n]
#pragma unroll
        for (int nt = 0; nt < 8; ++nt) {
            FragBF bf; load_b_frag(bf, &ldsVT[(nt * 16 + r) * ATS], hi); // B[n][e]
            acc[nt] = __builtin_amdgcn_wmma_f32_16x16x32_bf16(
                false, af.v, false, bf.v, (short)0, acc[nt], false, false);
        }
    }
#pragma unroll
    for (int j = 0; j < 8; ++j) {
        int d = wave * 16 + j + hi * 8;
#pragma unroll
        for (int nt = 0; nt < 8; ++nt) {
            int e = nt * 16 + r;
            kvT[(size_t)s * (D_MODEL * D_MODEL) + (size_t)e * D_MODEL + d] = f2bf(acc[nt][j]);
        }
    }
}

// ---------------- small glue kernels ----------------
__global__ void f32_to_bf16_k(const float* __restrict__ src,
                              unsigned short* __restrict__ dst, int n) {
    int i = blockIdx.x * 256 + threadIdx.x;
    if (i < n) dst[i] = f2bf(src[i]);
}

__global__ void embed_kernel(const int* __restrict__ src, const float* __restrict__ emb,
                             float* __restrict__ xf, unsigned short* __restrict__ xb) {
    int idx = blockIdx.x * 256 + threadIdx.x;           // M*128 threads
    int m = idx >> 7, d = idx & 127;
    int s = m >> 9, b = m & 511;
    int tok = src[b * SS + s];
    float v = emb[tok * D_MODEL + d] * 11.313708498984761f;   // sqrt(128)
    int i = d >> 1;
    float freq = __expf((float)i * -0.14391156831212787f);    // -ln(10000)/64
    float ang = (float)s * freq;
    v += (d & 1) ? __cosf(ang) : __sinf(ang);
    xf[idx] = v;
    xb[idx] = f2bf(v);
}

__global__ void ksum_kernel(const unsigned short* __restrict__ Kb, float* __restrict__ ks) {
    int s = blockIdx.x, d = threadIdx.x;                // 128 threads
    float acc = 0.0f;
    for (int n = 0; n < BB; ++n) acc += bf2f(Kb[((size_t)s * BB + n) * D_MODEL + d]);
    ks[s * D_MODEL + d] = acc;
}

__global__ void z_kernel(const unsigned short* __restrict__ Qb,
                         const float* __restrict__ ks, float* __restrict__ z) {
    int m = blockIdx.x * 256 + threadIdx.x;             // M threads
    int s = m >> 9;
    const unsigned short* q = Qb + (size_t)m * D_MODEL;
    const float* kr = ks + s * D_MODEL;
    float acc = 0.0f;
    for (int d = 0; d < D_MODEL; ++d) acc += bf2f(q[d]) * kr[d];
    z[m] = 1.0f / (acc + EPS_F);
}

__global__ void pool_fc_kernel(const float* __restrict__ xf, const float* __restrict__ fcw,
                               const float* __restrict__ fcb, float* __restrict__ out) {
    __shared__ float pooled[D_MODEL];
    int b = blockIdx.x, d = threadIdx.x;                // 128 threads
    float acc = 0.0f;
    for (int s = 0; s < SS; ++s) acc += xf[((size_t)s * BB + b) * D_MODEL + d];
    pooled[d] = acc * (1.0f / (float)SS);
    __syncthreads();
    if (d < 10) {
        float o = fcb[d];
        const float* w = fcw + d * D_MODEL;
        for (int k = 0; k < D_MODEL; ++k) o += pooled[k] * w[k];
        out[b * 10 + d] = o;
    }
}

// =====================================================================
extern "C" void kernel_launch(void* const* d_in, const int* in_sizes, int n_in,
                              void* d_out, int out_size, void* d_ws, size_t ws_size,
                              hipStream_t stream) {
    const int*   src   = (const int*)  d_in[0];
    const float* emb   = (const float*)d_in[1];
    const float* wq    = (const float*)d_in[2];
    const float* bq    = (const float*)d_in[3];
    const float* wk    = (const float*)d_in[4];
    const float* bk    = (const float*)d_in[5];
    const float* wv    = (const float*)d_in[6];
    const float* bv    = (const float*)d_in[7];
    const float* w1    = (const float*)d_in[8];
    const float* b1    = (const float*)d_in[9];
    const float* w2    = (const float*)d_in[10];
    const float* b2    = (const float*)d_in[11];
    const float* g1    = (const float*)d_in[12];
    const float* beta1 = (const float*)d_in[13];
    const float* g2    = (const float*)d_in[14];
    const float* beta2 = (const float*)d_in[15];
    const float* fcw   = (const float*)d_in[16];
    const float* fcb   = (const float*)d_in[17];
    (void)in_sizes; (void)n_in; (void)out_size; (void)ws_size;

    // ---- workspace carve ----
    char* p = (char*)d_ws;
    auto take = [&](size_t bytes) { char* q = p; p += (bytes + 255) & ~(size_t)255; return q; };
    float*          xf   = (float*)         take((size_t)MROWS * D_MODEL * 4);
    unsigned short* xb   = (unsigned short*)take((size_t)MROWS * D_MODEL * 2);
    unsigned short* qb   = (unsigned short*)take((size_t)MROWS * D_MODEL * 2);
    unsigned short* kb   = (unsigned short*)take((size_t)MROWS * D_MODEL * 2);
    unsigned short* vb   = (unsigned short*)take((size_t)MROWS * D_MODEL * 2);
    unsigned short* hb   = (unsigned short*)take((size_t)MROWS * D_FF * 2);
    unsigned short* kvT  = (unsigned short*)take((size_t)SS * D_MODEL * D_MODEL * 2);
    float*          ksum = (float*)         take((size_t)SS * D_MODEL * 4);
    float*          zb   = (float*)         take((size_t)MROWS * 4);
    unsigned short* wqb  = (unsigned short*)take((size_t)NUM_L * D_MODEL * D_MODEL * 2);
    unsigned short* wkb  = (unsigned short*)take((size_t)NUM_L * D_MODEL * D_MODEL * 2);
    unsigned short* wvb  = (unsigned short*)take((size_t)NUM_L * D_MODEL * D_MODEL * 2);
    unsigned short* w1b  = (unsigned short*)take((size_t)NUM_L * D_FF * D_MODEL * 2);
    unsigned short* w2b  = (unsigned short*)take((size_t)NUM_L * D_MODEL * D_FF * 2);

    // ---- weights -> bf16 (tiny) ----
    const int NWQ = NUM_L * D_MODEL * D_MODEL;   // 32768
    const int NW1 = NUM_L * D_FF * D_MODEL;      // 131072
    f32_to_bf16_k<<<(NWQ + 255) / 256, 256, 0, stream>>>(wq, wqb, NWQ);
    f32_to_bf16_k<<<(NWQ + 255) / 256, 256, 0, stream>>>(wk, wkb, NWQ);
    f32_to_bf16_k<<<(NWQ + 255) / 256, 256, 0, stream>>>(wv, wvb, NWQ);
    f32_to_bf16_k<<<(NW1 + 255) / 256, 256, 0, stream>>>(w1, w1b, NW1);
    f32_to_bf16_k<<<(NW1 + 255) / 256, 256, 0, stream>>>(w2, w2b, NW1);

    // ---- embedding + positional encoding ----
    embed_kernel<<<(MROWS * D_MODEL) / 256, 256, 0, stream>>>(src, emb, xf, xb);

    const int GB = MROWS / 128;   // 1024 row blocks
    for (int l = 0; l < NUM_L; ++l) {
        const unsigned short* wq_l = wqb + (size_t)l * D_MODEL * D_MODEL;
        const unsigned short* wk_l = wkb + (size_t)l * D_MODEL * D_MODEL;
        const unsigned short* wv_l = wvb + (size_t)l * D_MODEL * D_MODEL;
        const unsigned short* w1_l = w1b + (size_t)l * D_FF * D_MODEL;
        const unsigned short* w2_l = w2b + (size_t)l * D_MODEL * D_FF;

        // Q,K = relu(xW^T+b)+eps ; V = xW^T+b
        gemm_bias_act<<<dim3(GB, 1), 256, 0, stream>>>(xb, wq_l, bq + l * D_MODEL, qb, D_MODEL, D_MODEL, 1);
        gemm_bias_act<<<dim3(GB, 1), 256, 0, stream>>>(xb, wk_l, bk + l * D_MODEL, kb, D_MODEL, D_MODEL, 1);
        gemm_bias_act<<<dim3(GB, 1), 256, 0, stream>>>(xb, wv_l, bv + l * D_MODEL, vb, D_MODEL, D_MODEL, 0);

        ksum_kernel<<<SS, D_MODEL, 0, stream>>>(kb, ksum);
        kv_kernel<<<SS, 256, 0, stream>>>(kb, vb, kvT);
        z_kernel<<<MROWS / 256, 256, 0, stream>>>(qb, ksum, zb);

        // attn = Z*(Q@KV); x = LN(x+attn)   (per-s B matrix, Z row scale, no bias)
        gemm_ln<<<dim3(GB, 1), 256, 0, stream>>>(qb, kvT, nullptr, zb,
                                                 g1 + l * D_MODEL, beta1 + l * D_MODEL,
                                                 xf, xb, D_MODEL, 1);
        // h = relu(xW1^T+b1)
        gemm_bias_act<<<dim3(GB, D_FF / 128), 256, 0, stream>>>(xb, w1_l, b1 + l * D_FF, hb, D_MODEL, D_FF, 2);
        // x = LN(x + hW2^T + b2)
        gemm_ln<<<dim3(GB, 1), 256, 0, stream>>>(hb, w2_l, b2 + l * D_MODEL, nullptr,
                                                 g2 + l * D_MODEL, beta2 + l * D_MODEL,
                                                 xf, xb, D_FF, 0);
    }

    pool_fc_kernel<<<BB, D_MODEL, 0, stream>>>(xf, fcw, fcb, (float*)d_out);
}
